// DecoderLayer_23072564314620
// MI455X (gfx1250) — matrix operand order repaired and checked by reading, compiled, not run on hardware
//
#include <hip/hip_runtime.h>
#include <hip/hip_bf16.h>
#include <math.h>

// Problem dims (fixed by reference)
#define SQ_  2048
#define SK_  3072
#define TT_  4096
#define DM_  2048
#define NH_  16
#define NKV_ 8
#define HD_  128
#define FF_  6144

typedef __bf16 bf16;
typedef __attribute__((ext_vector_type(16))) __bf16 v16bf;
typedef __attribute__((ext_vector_type(8)))  __bf16 v8bf;
typedef __attribute__((ext_vector_type(8)))  float  v8f;

// ------------------------------------------------------------------
// CDNA5 async global->LDS copy (16B per lane), with sync fallback
// Builtin signature (from probe diagnostics): param0 is
// 'int __vector_size__(16) __device__ *'  i.e. v4i in addrspace(1),
// param1 the LDS-side v4i pointer, then imm offset, imm cpol.
// ------------------------------------------------------------------
#if __has_builtin(__builtin_amdgcn_global_load_async_to_lds_b128)
#define HAVE_ASYNC_LDS 1
typedef int v4i_vs __attribute__((vector_size(16)));
typedef v4i_vs __attribute__((address_space(1)))* as1_v4i;
typedef v4i_vs __attribute__((address_space(3)))* as3_v4i;
#endif

__device__ __forceinline__ void copy16_g2l(bf16* l, const bf16* g) {
#ifdef HAVE_ASYNC_LDS
  __builtin_amdgcn_global_load_async_to_lds_b128(
      (as1_v4i)const_cast<bf16*>(g), (as3_v4i)l, 0, 0);
#else
  *reinterpret_cast<uint4*>(l) = *reinterpret_cast<const uint4*>(g);
#endif
}

__device__ __forceinline__ void wait_async_all() {
#ifdef HAVE_ASYNC_LDS
#if __has_builtin(__builtin_amdgcn_s_wait_asynccnt)
  __builtin_amdgcn_s_wait_asynccnt(0);
#else
  asm volatile("s_wait_asynccnt 0x0" ::: "memory");
#endif
#endif
}

// ------------------------------------------------------------------
// Elementwise kernels
// ------------------------------------------------------------------
__global__ void f32_to_bf16_kernel(const float* __restrict__ in,
                                   bf16* __restrict__ out, long n) {
  long i = (long)blockIdx.x * blockDim.x + threadIdx.x;
  if (i < n) out[i] = (bf16)in[i];
}

__global__ void add_f32_kernel(const float* __restrict__ a,
                               const float* __restrict__ b,
                               float* __restrict__ out, long n) {
  long i = (long)blockIdx.x * blockDim.x + threadIdx.x;
  if (i < n) out[i] = a[i] + b[i];
}

__global__ void silu_mul_kernel(const float* __restrict__ g,
                                const float* __restrict__ u,
                                bf16* __restrict__ out, long n) {
  long i = (long)blockIdx.x * blockDim.x + threadIdx.x;
  if (i < n) {
    float x = g[i];
    float s = x / (1.0f + __expf(-x));
    out[i] = (bf16)(s * u[i]);
  }
}

// Tiled transpose-convert: in fp32 [K][N] -> out bf16 [N][K]
// grid (N/32, K/32), block (32,8)
__global__ __launch_bounds__(256)
void transpose_f32_to_bf16_kernel(const float* __restrict__ in,
                                  bf16* __restrict__ out, int K, int N) {
  __shared__ float tile[32][33];
  int n0 = blockIdx.x * 32, k0 = blockIdx.y * 32;
  for (int i = threadIdx.y; i < 32; i += 8)
    tile[i][threadIdx.x] = in[(size_t)(k0 + i) * N + n0 + threadIdx.x];
  __syncthreads();
  for (int i = threadIdx.y; i < 32; i += 8)
    out[(size_t)(n0 + i) * K + k0 + threadIdx.x] = (bf16)tile[threadIdx.x][i];
}

// ------------------------------------------------------------------
// Row RMSNorm, fp32 in -> bf16 out, weight applied
// ------------------------------------------------------------------
__global__ __launch_bounds__(256)
void rmsnorm_kernel(const float* __restrict__ x, const float* __restrict__ w,
                    bf16* __restrict__ y, int ncols) {
  int row = blockIdx.x, tid = threadIdx.x;
  const float* xr = x + (size_t)row * ncols;
  float ss = 0.f;
  for (int c = tid; c < ncols; c += 256) { float v = xr[c]; ss += v * v; }
  __shared__ float red[256];
  red[tid] = ss; __syncthreads();
  for (int off = 128; off > 0; off >>= 1) {
    if (tid < off) red[tid] += red[tid + off];
    __syncthreads();
  }
  float rms = rsqrtf(red[0] / (float)ncols + 1e-6f);
  bf16* yr = y + (size_t)row * ncols;
  for (int c = tid; c < ncols; c += 256) yr[c] = (bf16)(xr[c] * rms * w[c]);
}

// ------------------------------------------------------------------
// Per-head RMSNorm + RoPE + relayout to [head][S][128] bf16
// grid (S, nheads), block 64: thread d handles pair (d, d+64)
// ------------------------------------------------------------------
__global__ __launch_bounds__(64)
void qk_rope_kernel(const float* __restrict__ x,   // [S, nheads*128]
                    const float* __restrict__ w,   // [128]
                    const int* __restrict__ pos,   // [S]
                    bf16* __restrict__ out,        // [nheads][S][128]
                    int S, int nheads) {
  int s = blockIdx.x, hh = blockIdx.y, d = threadIdx.x;
  const float* xp = x + ((size_t)s * nheads + hh) * HD_;
  float x1 = xp[d], x2 = xp[d + 64];
  __shared__ float red[64];
  red[d] = x1 * x1 + x2 * x2;
  __syncthreads();
  for (int off = 32; off > 0; off >>= 1) {
    if (d < off) red[d] += red[d + off];
    __syncthreads();
  }
  float rms = rsqrtf(red[0] / (float)HD_ + 1e-6f);
  float xn1 = x1 * rms * w[d];
  float xn2 = x2 * rms * w[d + 64];
  float inv = powf(1000000.0f, -(float)(2 * d) / (float)HD_);
  float f = (float)pos[s] * inv;
  float c = __cosf(f), si = __sinf(f);
  bf16* op = out + ((size_t)hh * S + s) * HD_;
  op[d]      = (bf16)(xn1 * c - xn2 * si);
  op[d + 64] = (bf16)(xn2 * c + xn1 * si);
}

// v: [S, nheads*128] fp32 -> transposed heads [nheads][128][S] bf16
// grid (S/32, HD/32, nheads), block (32,8)
__global__ __launch_bounds__(256)
void split_v_t_kernel(const float* __restrict__ v, bf16* __restrict__ out,
                      int S, int nheads) {
  __shared__ float tile[32][33];
  int hh = blockIdx.z;
  int s0 = blockIdx.x * 32, d0 = blockIdx.y * 32;
  for (int i = threadIdx.y; i < 32; i += 8)
    tile[i][threadIdx.x] =
        v[(size_t)(s0 + i) * (nheads * HD_) + (size_t)hh * HD_ + d0 + threadIdx.x];
  __syncthreads();
  bf16* o = out + (size_t)hh * HD_ * S;
  for (int i = threadIdx.y; i < 32; i += 8)
    o[(size_t)(d0 + i) * S + s0 + threadIdx.x] = (bf16)tile[threadIdx.x][i];
}

// ------------------------------------------------------------------
// Masked softmax over SK=3072 per query row.
// mask[q][k] = causal_mask[hs_idxs[q]][key_idxs[k]]
// ------------------------------------------------------------------
__global__ __launch_bounds__(256)
void softmax_kernel(const float* __restrict__ S,
                    const float* __restrict__ mask,   // [TT, TT]
                    const int* __restrict__ hs_idxs,
                    const int* __restrict__ key_idxs,
                    bf16* __restrict__ P, float scale) {
  const int NPER = SK_ / 256;  // 12
  int q = blockIdx.x, tid = threadIdx.x;
  const float* srow = S + (size_t)q * SK_;
  const float* mrow = mask + (size_t)hs_idxs[q] * TT_;
  float vals[NPER];
  float mx = -3.0e38f;
  for (int i = 0; i < NPER; ++i) {
    int k = tid + i * 256;
    float v = srow[k] * scale + mrow[key_idxs[k]];
    vals[i] = v;
    mx = fmaxf(mx, v);
  }
  __shared__ float red[256];
  red[tid] = mx; __syncthreads();
  for (int off = 128; off > 0; off >>= 1) {
    if (tid < off) red[tid] = fmaxf(red[tid], red[tid + off]);
    __syncthreads();
  }
  mx = red[0]; __syncthreads();
  float sum = 0.f;
  for (int i = 0; i < NPER; ++i) {
    vals[i] = __expf(vals[i] - mx);
    sum += vals[i];
  }
  red[tid] = sum; __syncthreads();
  for (int off = 128; off > 0; off >>= 1) {
    if (tid < off) red[tid] += red[tid + off];
    __syncthreads();
  }
  float invs = 1.0f / red[0];
  bf16* prow = P + (size_t)q * SK_;
  for (int i = 0; i < NPER; ++i) prow[tid + i * 256] = (bf16)(vals[i] * invs);
}

// ------------------------------------------------------------------
// WMMA bf16 GEMM (NT): C[M,N] (fp32) = A[M,K] @ B[N,K]^T
// Both operands row-major along K. Block tile 128x128, BK=32,
// 256 threads = 8 waves (2x4), each wave 4x2 v_wmma_f32_16x16x32_bf16.
// Double-buffered LDS with async global->LDS copies (ASYNCcnt).
// M,N multiples of 128; K multiple of 32; lda/ldb multiples of 8.
// ------------------------------------------------------------------
__global__ __launch_bounds__(256)
void gemm_nt_wmma_kernel(const bf16* __restrict__ A, const bf16* __restrict__ B,
                         float* __restrict__ C, int M, int N, int K,
                         int lda, int ldb, int ldc) {
  // 128 rows x 32 k-halfs, padded to 40 halfs (80B rows, 16B aligned)
  __shared__ __align__(16) bf16 As[2][128 * 40];
  __shared__ __align__(16) bf16 Bs[2][128 * 40];

  const int tid  = threadIdx.x;
  const int wid  = tid >> 5;
  const int lane = tid & 31;
  const int wm   = wid >> 2;      // 0..1
  const int wn   = wid & 3;       // 0..3
  const int h    = lane >> 4;     // half-select (0/1)
  const int r    = lane & 15;     // row/col within tile
  const int blockM = blockIdx.y * 128;
  const int blockN = blockIdx.x * 128;

  // straight-line tile stage: 2 x 16B per thread per operand
  const int row0 = tid >> 2, kc0 = (tid & 3) << 3;          // chunk tid
  const int row1 = (tid + 256) >> 2, kc1 = kc0;             // chunk tid+256

  auto stage = [&](int buf, int k0) {
    copy16_g2l(&As[buf][row0 * 40 + kc0],
               A + (size_t)(blockM + row0) * lda + (k0 + kc0));
    copy16_g2l(&As[buf][row1 * 40 + kc1],
               A + (size_t)(blockM + row1) * lda + (k0 + kc1));
    copy16_g2l(&Bs[buf][row0 * 40 + kc0],
               B + (size_t)(blockN + row0) * ldb + (k0 + kc0));
    copy16_g2l(&Bs[buf][row1 * 40 + kc1],
               B + (size_t)(blockN + row1) * ldb + (k0 + kc1));
  };

  v8f zero = {0.f, 0.f, 0.f, 0.f, 0.f, 0.f, 0.f, 0.f};
  v8f acc[4][2];
#pragma unroll
  for (int i = 0; i < 4; ++i)
#pragma unroll
    for (int j = 0; j < 2; ++j) acc[i][j] = zero;

  const int nk = K >> 5;
  stage(0, 0);
  wait_async_all();
  __syncthreads();

  for (int t = 0; t < nk; ++t) {
    const int cur = t & 1;
    if (t + 1 < nk) stage(cur ^ 1, (t + 1) << 5);   // prefetch next tile

    // ---- fragments per ISA bf16 layouts ----
    v16bf afrag[4], bfrag[2];
#pragma unroll
    for (int i = 0; i < 4; ++i) {
      int rr = wm * 64 + i * 16 + r;
      v8bf lo = *reinterpret_cast<const v8bf*>(&As[cur][rr * 40 + h * 8]);
      v8bf hi = *reinterpret_cast<const v8bf*>(&As[cur][rr * 40 + 16 + h * 8]);
      afrag[i] = __builtin_shufflevector(lo, hi, 0, 1, 2, 3, 4, 5, 6, 7,
                                         8, 9, 10, 11, 12, 13, 14, 15);
    }
#pragma unroll
    for (int j = 0; j < 2; ++j) {
      int cc = wn * 32 + j * 16 + r;
      v8bf lo = *reinterpret_cast<const v8bf*>(&Bs[cur][cc * 40 + h * 8]);
      v8bf hi = *reinterpret_cast<const v8bf*>(&Bs[cur][cc * 40 + 16 + h * 8]);
      bfrag[j] = __builtin_shufflevector(lo, hi, 0, 1, 2, 3, 4, 5, 6, 7,
                                         8, 9, 10, 11, 12, 13, 14, 15);
    }

#pragma unroll
    for (int i = 0; i < 4; ++i)
#pragma unroll
      for (int j = 0; j < 2; ++j)
        acc[i][j] = __builtin_amdgcn_wmma_f32_16x16x32_bf16(
            false, afrag[i], false, bfrag[j], (short)0, acc[i][j],
            false, false);

    wait_async_all();   // prefetched tile resident in LDS
    __syncthreads();    // all waves done reading 'cur', prefetch visible
  }

  // ---- epilogue: C/D layout => VGPR e holds (M = e + 8h, N = r) ----
#pragma unroll
  for (int i = 0; i < 4; ++i) {
#pragma unroll
    for (int j = 0; j < 2; ++j) {
      int gm = blockM + wm * 64 + i * 16 + 8 * h;
      int gn = blockN + wn * 32 + j * 16 + r;
      float* cp = C + (size_t)gm * ldc + gn;
#pragma unroll
      for (int e = 0; e < 8; ++e) cp[(size_t)e * ldc] = acc[i][j][e];
    }
  }
}

// ------------------------------------------------------------------
// Host-side orchestration
// ------------------------------------------------------------------
static void* wsalloc(char** p, size_t bytes) {
  void* r = (void*)*p;
  *p += (bytes + 255) & ~(size_t)255;
  return r;
}

static inline int ceil_div_i(long n, int d) { return (int)((n + d - 1) / d); }

extern "C" void kernel_launch(void* const* d_in, const int* in_sizes, int n_in,
                              void* d_out, int out_size, void* d_ws, size_t ws_size,
                              hipStream_t stream) {
  (void)in_sizes; (void)n_in; (void)out_size; (void)ws_size;
  const float* hidden_states = (const float*)d_in[0];   // [SQ, DM]
  const float* kv_hidden     = (const float*)d_in[1];   // [SK, DM]
  const float* causal_mask   = (const float*)d_in[2];   // [TT, TT]
  const float* w_q      = (const float*)d_in[3];        // [DM, NH*HD]
  const float* w_k      = (const float*)d_in[4];        // [DM, NKV*HD]
  const float* w_v      = (const float*)d_in[5];        // [DM, NKV*HD]
  const float* w_o      = (const float*)d_in[6];        // [NH*HD, DM]
  const float* q_norm_w = (const float*)d_in[7];        // [HD]
  const float* k_norm_w = (const float*)d_in[8];        // [HD]
  const float* ln1_w    = (const float*)d_in[9];        // [DM]
  const float* ln2_w    = (const float*)d_in[10];       // [DM]
  const float* w_gate   = (const float*)d_in[11];       // [DM, FF]
  const float* w_up     = (const float*)d_in[12];       // [DM, FF]
  const float* w_down   = (const float*)d_in[13];       // [FF, DM]
  const int* positions    = (const int*)d_in[14];       // [SQ]
  const int* kv_positions = (const int*)d_in[15];       // [SK]
  const int* hs_idxs      = (const int*)d_in[16];       // [SQ]
  const int* key_idxs     = (const int*)d_in[17];       // [SK]
  float* out = (float*)d_out;                           // [SQ, DM]

  char* wp = (char*)d_ws;
  // transposed bf16 weights: [N][K] layouts for the NT GEMM
  bf16* wq_t  = (bf16*)wsalloc(&wp, (size_t)DM_ * NH_ * HD_ * 2);   // [NH*HD][DM]
  bf16* wk_t  = (bf16*)wsalloc(&wp, (size_t)DM_ * NKV_ * HD_ * 2);  // [NKV*HD][DM]
  bf16* wv_t  = (bf16*)wsalloc(&wp, (size_t)DM_ * NKV_ * HD_ * 2);  // [NKV*HD][DM]
  bf16* wo_t  = (bf16*)wsalloc(&wp, (size_t)NH_ * HD_ * DM_ * 2);   // [DM][NH*HD]
  bf16* wg_t  = (bf16*)wsalloc(&wp, (size_t)DM_ * FF_ * 2);         // [FF][DM]
  bf16* wu_t  = (bf16*)wsalloc(&wp, (size_t)DM_ * FF_ * 2);         // [FF][DM]
  bf16* wd_t  = (bf16*)wsalloc(&wp, (size_t)FF_ * DM_ * 2);         // [DM][FF]
  bf16* h_bf   = (bf16*)wsalloc(&wp, (size_t)SQ_ * DM_ * 2);
  bf16* hk_bf  = (bf16*)wsalloc(&wp, (size_t)SK_ * DM_ * 2);
  float* q_f32 = (float*)wsalloc(&wp, (size_t)SQ_ * NH_ * HD_ * 4);
  float* k_f32 = (float*)wsalloc(&wp, (size_t)SK_ * NKV_ * HD_ * 4);
  float* v_f32 = (float*)wsalloc(&wp, (size_t)SK_ * NKV_ * HD_ * 4);
  bf16* q_bf   = (bf16*)wsalloc(&wp, (size_t)NH_ * SQ_ * HD_ * 2);  // [NH][SQ][HD]
  bf16* k_bf   = (bf16*)wsalloc(&wp, (size_t)NKV_ * SK_ * HD_ * 2); // [NKV][SK][HD]
  bf16* vt_bf  = (bf16*)wsalloc(&wp, (size_t)NKV_ * HD_ * SK_ * 2); // [NKV][HD][SK]
  float* scores = (float*)wsalloc(&wp, (size_t)SQ_ * SK_ * 4);
  bf16*  probs  = (bf16*)wsalloc(&wp, (size_t)SQ_ * SK_ * 2);
  float* ctx_f32 = (float*)wsalloc(&wp, (size_t)SQ_ * NH_ * HD_ * 4);
  bf16*  ctx_bf  = (bf16*)wsalloc(&wp, (size_t)SQ_ * NH_ * HD_ * 2);
  float* attn_f32 = (float*)wsalloc(&wp, (size_t)SQ_ * DM_ * 4);
  float* hidden_f32 = (float*)wsalloc(&wp, (size_t)SQ_ * DM_ * 4);
  bf16*  h2_bf   = (bf16*)wsalloc(&wp, (size_t)SQ_ * DM_ * 2);
  float* gate_f32 = (float*)wsalloc(&wp, (size_t)SQ_ * FF_ * 4);
  float* up_f32   = (float*)wsalloc(&wp, (size_t)SQ_ * FF_ * 4);
  bf16*  act_bf   = (bf16*)wsalloc(&wp, (size_t)SQ_ * FF_ * 2);
  float* mlp_f32  = (float*)wsalloc(&wp, (size_t)SQ_ * DM_ * 4);

  auto transp = [&](const float* src, bf16* dst, int K, int N) {
    transpose_f32_to_bf16_kernel<<<dim3(N / 32, K / 32), dim3(32, 8), 0, stream>>>(
        src, dst, K, N);
  };
  auto cvt = [&](const float* src, bf16* dst, long n) {
    f32_to_bf16_kernel<<<ceil_div_i(n, 256), 256, 0, stream>>>(src, dst, n);
  };

  // 1) weights -> transposed bf16
  transp(w_q, wq_t, DM_, NH_ * HD_);
  transp(w_k, wk_t, DM_, NKV_ * HD_);
  transp(w_v, wv_t, DM_, NKV_ * HD_);
  transp(w_o, wo_t, NH_ * HD_, DM_);
  transp(w_gate, wg_t, DM_, FF_);
  transp(w_up, wu_t, DM_, FF_);
  transp(w_down, wd_t, FF_, DM_);

  // 2) input RMSNorm -> bf16
  rmsnorm_kernel<<<SQ_, 256, 0, stream>>>(hidden_states, ln1_w, h_bf, DM_);
  rmsnorm_kernel<<<SK_, 256, 0, stream>>>(kv_hidden, ln1_w, hk_bf, DM_);

  // 3) QKV projections (WMMA, NT against transposed weights)
  gemm_nt_wmma_kernel<<<dim3((NH_ * HD_) / 128, SQ_ / 128), 256, 0, stream>>>(
      h_bf, wq_t, q_f32, SQ_, NH_ * HD_, DM_, DM_, DM_, NH_ * HD_);
  gemm_nt_wmma_kernel<<<dim3((NKV_ * HD_) / 128, SK_ / 128), 256, 0, stream>>>(
      hk_bf, wk_t, k_f32, SK_, NKV_ * HD_, DM_, DM_, DM_, NKV_ * HD_);
  gemm_nt_wmma_kernel<<<dim3((NKV_ * HD_) / 128, SK_ / 128), 256, 0, stream>>>(
      hk_bf, wv_t, v_f32, SK_, NKV_ * HD_, DM_, DM_, DM_, NKV_ * HD_);

  // 4) per-head q/k RMSNorm + RoPE, head-major relayout; v transposed split
  qk_rope_kernel<<<dim3(SQ_, NH_), 64, 0, stream>>>(q_f32, q_norm_w, positions, q_bf, SQ_, NH_);
  qk_rope_kernel<<<dim3(SK_, NKV_), 64, 0, stream>>>(k_f32, k_norm_w, kv_positions, k_bf, SK_, NKV_);
  split_v_t_kernel<<<dim3(SK_ / 32, HD_ / 32, NKV_), dim3(32, 8), 0, stream>>>(
      v_f32, vt_bf, SK_, NKV_);

  // 5) attention, head by head (GQA: kv head = h/2)
  const float scale = 1.0f / sqrtf((float)HD_);
  for (int hh = 0; hh < NH_; ++hh) {
    const bf16* qh  = q_bf + (size_t)hh * SQ_ * HD_;
    const bf16* kh  = k_bf + (size_t)(hh / 2) * SK_ * HD_;
    const bf16* vth = vt_bf + (size_t)(hh / 2) * HD_ * SK_;
    // scores = q @ k^T  (NT: both [S][128] row-major)
    gemm_nt_wmma_kernel<<<dim3(SK_ / 128, SQ_ / 128), 256, 0, stream>>>(
        qh, kh, scores, SQ_, SK_, HD_, HD_, HD_, SK_);
    // masked softmax -> bf16 probs
    softmax_kernel<<<SQ_, 256, 0, stream>>>(scores, causal_mask, hs_idxs,
                                            key_idxs, probs, scale);
    // ctx = probs @ (v^T)^T  (NT: B = vt [HD][SK]) -> column slice of [SQ][NH*HD]
    gemm_nt_wmma_kernel<<<dim3(HD_ / 128, SQ_ / 128), 256, 0, stream>>>(
        probs, vth, ctx_f32 + (size_t)hh * HD_, SQ_, HD_, SK_, SK_, SK_, NH_ * HD_);
  }

  // 6) output projection + residual
  cvt(ctx_f32, ctx_bf, (long)SQ_ * NH_ * HD_);
  gemm_nt_wmma_kernel<<<dim3(DM_ / 128, SQ_ / 128), 256, 0, stream>>>(
      ctx_bf, wo_t, attn_f32, SQ_, DM_, NH_ * HD_, NH_ * HD_, NH_ * HD_, DM_);
  add_f32_kernel<<<ceil_div_i((long)SQ_ * DM_, 256), 256, 0, stream>>>(
      hidden_states, attn_f32, hidden_f32, (long)SQ_ * DM_);

  // 7) post-attention RMSNorm + SwiGLU MLP
  rmsnorm_kernel<<<SQ_, 256, 0, stream>>>(hidden_f32, ln2_w, h2_bf, DM_);
  gemm_nt_wmma_kernel<<<dim3(FF_ / 128, SQ_ / 128), 256, 0, stream>>>(
      h2_bf, wg_t, gate_f32, SQ_, FF_, DM_, DM_, DM_, FF_);
  gemm_nt_wmma_kernel<<<dim3(FF_ / 128, SQ_ / 128), 256, 0, stream>>>(
      h2_bf, wu_t, up_f32, SQ_, FF_, DM_, DM_, DM_, FF_);
  silu_mul_kernel<<<ceil_div_i((long)SQ_ * FF_, 256), 256, 0, stream>>>(
      gate_f32, up_f32, act_bf, (long)SQ_ * FF_);
  gemm_nt_wmma_kernel<<<dim3(DM_ / 128, SQ_ / 128), 256, 0, stream>>>(
      act_bf, wd_t, mlp_f32, SQ_, DM_, FF_, FF_, FF_, DM_);

  // 8) final residual -> output
  add_f32_kernel<<<ceil_div_i((long)SQ_ * DM_, 256), 256, 0, stream>>>(
      hidden_f32, mlp_f32, out, (long)SQ_ * DM_);
}
